// SlotAttention_4466765988488
// MI455X (gfx1250) — matrix operand order repaired and checked by reading, not compiled, not run on hardware
//
#include <hip/hip_runtime.h>
#include <hip/hip_bf16.h>

typedef __attribute__((ext_vector_type(16))) _Float16 v16h;
typedef __attribute__((ext_vector_type(8)))  float    v8f;

#define B_ 64
#define N_ 4096
#define D_ 256
#define S_ 8
#define ITERS_ 3
#define EPS_ 1e-5f

#define WMMA_F16(A, Bm, C) \
  __builtin_amdgcn_wmma_f32_16x16x32_f16(false, (A), false, (Bm), (short)0, (C), false, false)

// ---------------------------------------------------------------- helpers

static __device__ __forceinline__ float wave_reduce_sum(float v) {
#pragma unroll
  for (int m = 16; m >= 1; m >>= 1) v += __shfl_xor(v, m, 32);
  return v;
}

// A-matrix fragment (16x32 f16), CDNA5 layout:
// lane l<16: row M=l, K = kBase+{0..7,16..23}; lane l+16: row M=l, K = kBase+{8..15,24..31}
static __device__ __forceinline__ v16h load_a_frag(const _Float16* base, int ld,
                                                   int mBase, int kBase, int lane) {
  const int l = lane & 15, hi = lane >> 4;
  const _Float16* p0 = base + (long)(mBase + l) * ld + kBase + hi * 8;
  v16h a;
#pragma unroll
  for (int i = 0; i < 8; ++i) { a[i] = p0[i]; a[8 + i] = p0[16 + i]; }
  return a;
}

// B-matrix fragment (32x16 f16) from a row-major f16 buffer:
// B[k,n] = buf[(rBase+n)*ld + cBase + k]  (lane half h: K = {8h..8h+7, 16+8h..16+8h+7})
static __device__ __forceinline__ v16h load_b_rm(const _Float16* __restrict__ buf,
                                                 long rBase, long ld, int cBase,
                                                 int lane) {
  const int n = lane & 15, hi = lane >> 4;
  const _Float16* p = buf + (rBase + n) * ld + cBase + hi * 8;
  v16h b;
#pragma unroll
  for (int i = 0; i < 8; ++i) { b[i] = p[i]; b[8 + i] = p[16 + i]; }
  return b;
}

// ---------------------------------------------------------------- kernel 0
__global__ void f32_to_f16_kernel(const float* __restrict__ src,
                                  _Float16* __restrict__ dst, int n) {
  int i = blockIdx.x * blockDim.x + threadIdx.x;
  if (i < n) dst[i] = (_Float16)src[i];
}

// ---------------------------------------------------------------- kernel 1
// Fused LayerNorm(x) + K/V projections. K row-major f16, V transposed f16.
#define K1_ROWS 32
#define K1_LD 264

__global__ __launch_bounds__(256)
void ln_proj_kv_kernel(const float* __restrict__ x,
                       const float* __restrict__ g_in, const float* __restrict__ b_in,
                       const _Float16* __restrict__ wk16, const float* __restrict__ bk,
                       const _Float16* __restrict__ wv16, const float* __restrict__ bv,
                       _Float16* __restrict__ ko, _Float16* __restrict__ vT) {
  __shared__ _Float16 xn[K1_ROWS * K1_LD];
  const int tid = threadIdx.x, lane = tid & 31, wave = tid >> 5;
  const long rowBase = (long)blockIdx.x * K1_ROWS;

  // --- LayerNorm: each wave normalizes 4 rows into LDS (f16) ---
#pragma unroll
  for (int rr = 0; rr < 4; ++rr) {
    const int r = wave * 4 + rr;
    const float* xr = x + (rowBase + r) * D_;
    float xs[8], s = 0.f, s2 = 0.f;
#pragma unroll
    for (int t = 0; t < 8; ++t) {
      float v = xr[lane + 32 * t];
      xs[t] = v; s += v; s2 += v * v;
    }
    s = wave_reduce_sum(s); s2 = wave_reduce_sum(s2);
    const float mean = s * (1.f / D_);
    const float rstd = rsqrtf(s2 * (1.f / D_) - mean * mean + EPS_);
#pragma unroll
    for (int t = 0; t < 8; ++t) {
      const int d = lane + 32 * t;
      xn[r * K1_LD + d] = (_Float16)((xs[t] - mean) * rstd * g_in[d] + b_in[d]);
    }
  }
  __syncthreads();

  // wave -> (matrix, row-tile, col-range); A fragments fixed per wave
  const int mat = wave >> 2;
  const int rowT = ((wave >> 1) & 1) * 16;
  const int colBase = (wave & 1) * 8;
  const _Float16* W = mat ? wv16 : wk16;
  const float* bias = mat ? bv : bk;

  v16h aF[8];
#pragma unroll
  for (int kk = 0; kk < 8; ++kk)
    aF[kk] = load_a_frag(xn, K1_LD, rowT, kk * 32, lane);

  const int mOff = (lane >> 4) * 8;
  const long bIdx = rowBase >> 12;
  const long n0 = (rowBase & (N_ - 1)) + rowT + mOff;

  for (int t = 0; t < 8; ++t) {
    const int eBase = (colBase + t) * 16;
    v8f c0 = {}, c1 = {};  // even/odd K accumulators -> 2 independent WMMA chains
#pragma unroll
    for (int kk = 0; kk < 8; kk += 2) {
      v16h b0 = load_b_rm(W, eBase, D_, kk * 32, lane);
      c0 = WMMA_F16(aF[kk], b0, c0);
      v16h b1 = load_b_rm(W, eBase, D_, kk * 32 + 32, lane);
      c1 = WMMA_F16(aF[kk + 1], b1, c1);
    }
    const int e = eBase + (lane & 15);
    const float bb = bias[e];
    if (mat == 0) {
#pragma unroll
      for (int r = 0; r < 8; ++r)
        ko[(rowBase + rowT + mOff + r) * D_ + e] = (_Float16)(c0[r] + c1[r] + bb);
    } else {
      _Float16* p = vT + (bIdx * D_ + e) * N_ + n0;
#pragma unroll
      for (int r = 0; r < 8; ++r) p[r] = (_Float16)(c0[r] + c1[r] + bb);
    }
  }
}

// ---------------------------------------------------------------- kernel 2
__global__ void slots_init_kernel(const float* __restrict__ noise,
                                  const float* __restrict__ mu,
                                  const float* __restrict__ sig,
                                  float* __restrict__ slots) {
  int i = blockIdx.x * blockDim.x + threadIdx.x;
  if (i >= B_ * S_ * D_) return;
  int d = i & (D_ - 1);
  slots[i] = mu[d] + log1pf(expf(sig[d])) * noise[i];
}

// ---------------------------------------------------------------- kernel 3
// q = LN(slots)@wq^T + bq via WMMA -> f16 padded to 16 rows (rows 8..15 zero)
__global__ __launch_bounds__(256)
void q_proj_kernel(const float* __restrict__ slots,
                   const float* __restrict__ g_sl, const float* __restrict__ b_sl,
                   const _Float16* __restrict__ wq16, const float* __restrict__ bq,
                   _Float16* __restrict__ q16) {
  __shared__ _Float16 sn16[16 * 264];
  const int b = blockIdx.x, tid = threadIdx.x, lane = tid & 31, wave = tid >> 5;
  for (int i = tid; i < 8 * 264; i += 256) sn16[8 * 264 + i] = (_Float16)0.f;

  {  // LN of slot row `wave`
    const float* sr = slots + ((long)b * S_ + wave) * D_;
    float xs[8], sm = 0.f, s2 = 0.f;
#pragma unroll
    for (int t = 0; t < 8; ++t) {
      float v = sr[lane + 32 * t];
      xs[t] = v; sm += v; s2 += v * v;
    }
    sm = wave_reduce_sum(sm); s2 = wave_reduce_sum(s2);
    const float mean = sm * (1.f / D_);
    const float rstd = rsqrtf(s2 * (1.f / D_) - mean * mean + EPS_);
#pragma unroll
    for (int t = 0; t < 8; ++t) {
      const int d = lane + 32 * t;
      sn16[wave * 264 + d] = (_Float16)((xs[t] - mean) * rstd * g_sl[d] + b_sl[d]);
    }
  }
  __syncthreads();

  const int mOff = (lane >> 4) * 8, eL = lane & 15;
  for (int g = wave; g < 16; g += 8) {
    const int eBase = g * 16;
    v8f c0 = {}, c1 = {};
#pragma unroll
    for (int kk = 0; kk < 8; kk += 2) {
      v16h a0 = load_a_frag(sn16, 264, 0, kk * 32, lane);
      v16h b0 = load_b_rm(wq16, eBase, D_, kk * 32, lane);
      c0 = WMMA_F16(a0, b0, c0);
      v16h a1 = load_a_frag(sn16, 264, 0, kk * 32 + 32, lane);
      v16h b1 = load_b_rm(wq16, eBase, D_, kk * 32 + 32, lane);
      c1 = WMMA_F16(a1, b1, c1);
    }
    const int e = eBase + eL;
    const float bb = bq[e];
#pragma unroll
    for (int r = 0; r < 8; ++r) {  // lanes>=16 write the zero pad rows 8..15
      float vv = (lane < 16) ? (c0[r] + c1[r] + bb) : 0.f;
      q16[((long)b * 16 + mOff + r) * D_ + e] = (_Float16)vv;
    }
  }
}

// ---------------------------------------------------------------- kernel 4
#define NCHUNK 8
#define JCHUNK 128
#define ATT_LD 136

__global__ __launch_bounds__(256)
void attn_kernel(const _Float16* __restrict__ q16, const _Float16* __restrict__ kmat,
                 const _Float16* __restrict__ vT, float* __restrict__ partial) {
  __shared__ _Float16 att[16 * ATT_LD];
  const int b = blockIdx.x, cb = blockIdx.y;
  const int tid = threadIdx.x, lane = tid & 31, wave = tid >> 5;

  const _Float16* qsrc = q16 + (long)b * 16 * D_;
  v16h qa[8];
#pragma unroll
  for (int kk = 0; kk < 8; ++kk) qa[kk] = load_a_frag(qsrc, D_, 0, kk * 32, lane);

  const long nBase = (long)b * N_;
  const _Float16* vTb = vT + (long)b * D_ * N_;
  const int jEnd = (cb + 1) * (N_ / NCHUNK);
  const int d0 = wave * 16, d1 = (wave + 8) * 16;
  v8f a0e = {}, a0o = {}, a1e = {}, a1o = {};  // split accumulators (2 chains each)

  for (int jb = cb * (N_ / NCHUNK); jb < jEnd; jb += JCHUNK) {
    // ---- phase 1: logits tile + per-lane softmax over the 8 slot rows ----
    const int jt = jb + wave * 16;
    if (jb + JCHUNK < jEnd) {
      __builtin_prefetch(kmat + (nBase + jt + JCHUNK) * D_, 0, 1);
      __builtin_prefetch(vTb + (long)d0 * N_ + jb + JCHUNK, 0, 1);
    }
    v8f c0 = {}, c1 = {};
#pragma unroll
    for (int kk = 0; kk < 8; kk += 2) {
      v16h b0 = load_b_rm(kmat, nBase + jt, D_, kk * 32, lane);
      c0 = WMMA_F16(qa[kk], b0, c0);
      v16h b1 = load_b_rm(kmat, nBase + jt, D_, kk * 32 + 32, lane);
      c1 = WMMA_F16(qa[kk + 1], b1, c1);
    }
    float lg[8];
#pragma unroll
    for (int r = 0; r < 8; ++r) lg[r] = c0[r] + c1[r];
    float mx = lg[0];
#pragma unroll
    for (int r = 1; r < 8; ++r) mx = fmaxf(mx, lg[r]);
    float ex[8], sum = 0.f;
#pragma unroll
    for (int r = 0; r < 8; ++r) { ex[r] = __expf(lg[r] - mx); sum += ex[r]; }
    const float sel = (lane < 16) ? (1.f / sum) : 0.f;  // pad rows -> 0
    const int col = wave * 16 + (lane & 15);
    const int rOff = (lane >> 4) * 8;
#pragma unroll
    for (int r = 0; r < 8; ++r)
      att[(rOff + r) * ATT_LD + col] = (_Float16)(ex[r] * sel);
    __syncthreads();

    // ---- phase 2: updates += attn @ v ; each wave owns 2 d-tiles ----
#pragma unroll
    for (int ks = 0; ks < 4; ks += 2) {
      v16h a = load_a_frag(att, ATT_LD, 0, ks * 32, lane);
      const int jc = jb + ks * 32;
      v16h b0 = load_b_rm(vTb, d0, N_, jc, lane);
      a0e = WMMA_F16(a, b0, a0e);
      v16h b1 = load_b_rm(vTb, d1, N_, jc, lane);
      a1e = WMMA_F16(a, b1, a1e);
      v16h a2 = load_a_frag(att, ATT_LD, 0, ks * 32 + 32, lane);
      const int jc2 = jc + 32;
      v16h b2 = load_b_rm(vTb, d0, N_, jc2, lane);
      a0o = WMMA_F16(a2, b2, a0o);
      v16h b3 = load_b_rm(vTb, d1, N_, jc2, lane);
      a1o = WMMA_F16(a2, b3, a1o);
    }
    __syncthreads();
  }

  if (lane < 16) {  // rows 0..7 (valid slots) live on lanes 0..15
    float* p = partial + ((long)b * NCHUNK + cb) * (S_ * D_);
#pragma unroll
    for (int r = 0; r < 8; ++r) {
      p[r * D_ + d0 + lane] = a0e[r] + a0o[r];
      p[r * D_ + d1 + lane] = a1e[r] + a1o[r];
    }
  }
}

// ---------------------------------------------------------------- kernel 4b
__global__ void reduce_updates_kernel(const float* __restrict__ partial,
                                      float* __restrict__ updates) {
  int i = blockIdx.x * blockDim.x + threadIdx.x;
  if (i >= B_ * S_ * D_) return;
  int b = i / (S_ * D_), sd = i % (S_ * D_);
  float s = 0.f;
#pragma unroll
  for (int c = 0; c < NCHUNK; ++c)
    s += partial[((long)b * NCHUNK + c) * (S_ * D_) + sd];
  updates[i] = s;
}

// ---------------------------------------------------------------- kernel 5
// GRU + LN + MLP, all matmuls on WMMA. A wave computes the six gate tiles
// (gi_r/z/n, gh_r/z/n) for one (rows x e-tile) -> gate math is fragment-local.
__global__ __launch_bounds__(256)
void gru_mlp_kernel(float* __restrict__ slots, const float* __restrict__ updates,
                    const _Float16* __restrict__ wih16, const _Float16* __restrict__ whh16,
                    const float* __restrict__ b_ih, const float* __restrict__ b_hh,
                    const _Float16* __restrict__ mw116, const float* __restrict__ mb1,
                    const _Float16* __restrict__ mw216, const float* __restrict__ mb2,
                    const float* __restrict__ g_ml, const float* __restrict__ b_ml) {
  __shared__ _Float16 u16[16 * 264];
  __shared__ _Float16 h16[16 * 264];
  __shared__ _Float16 sl16[16 * 264];
  __shared__ _Float16 h116[16 * 264];
  __shared__ float hf[S_ * D_];
  __shared__ float snf[S_ * D_];
  const int b = blockIdx.x, tid = threadIdx.x, lane = tid & 31, wave = tid >> 5;
  const long base = (long)b * S_ * D_;

  for (int i = tid; i < S_ * D_; i += 256) {
    const int s = i >> 8, d = i & 255;
    float hv = slots[base + i], uv = updates[base + i];
    hf[i] = hv;
    h16[s * 264 + d] = (_Float16)hv;
    u16[s * 264 + d] = (_Float16)uv;
  }
  for (int i = tid; i < 8 * 264; i += 256) {  // zero pad rows 8..15
    u16[8 * 264 + i] = (_Float16)0.f;
    h16[8 * 264 + i] = (_Float16)0.f;
    sl16[8 * 264 + i] = (_Float16)0.f;
    h116[8 * 264 + i] = (_Float16)0.f;
  }
  __syncthreads();

  const int eL = lane & 15;

  // ---- GRU gates via WMMA ----
  for (int g = wave; g < 16; g += 8) {
    const int eBase = g * 16;
    v8f gir = {}, giz = {}, gin = {}, ghr = {}, ghz = {}, ghn = {};
#pragma unroll
    for (int kk = 0; kk < 8; ++kk) {
      const int kb = kk * 32;
      v16h au = load_a_frag(u16, 264, 0, kb, lane);
      v16h ah = load_a_frag(h16, 264, 0, kb, lane);
      v16h bw;
      bw = load_b_rm(wih16, eBase, D_, kb, lane);        gir = WMMA_F16(au, bw, gir);
      bw = load_b_rm(wih16, 256 + eBase, D_, kb, lane);  giz = WMMA_F16(au, bw, giz);
      bw = load_b_rm(wih16, 512 + eBase, D_, kb, lane);  gin = WMMA_F16(au, bw, gin);
      bw = load_b_rm(whh16, eBase, D_, kb, lane);        ghr = WMMA_F16(ah, bw, ghr);
      bw = load_b_rm(whh16, 256 + eBase, D_, kb, lane);  ghz = WMMA_F16(ah, bw, ghz);
      bw = load_b_rm(whh16, 512 + eBase, D_, kb, lane);  ghn = WMMA_F16(ah, bw, ghn);
    }
    const int e = eBase + eL;
    const float bir = b_ih[e], biz = b_ih[256 + e], bin = b_ih[512 + e];
    const float bhr = b_hh[e], bhz = b_hh[256 + e], bhn = b_hh[512 + e];
    if (lane < 16) {  // valid slot rows 0..7
#pragma unroll
      for (int r = 0; r < 8; ++r) {
        float rg = 1.f / (1.f + expf(-((gir[r] + bir) + (ghr[r] + bhr))));
        float zg = 1.f / (1.f + expf(-((giz[r] + biz) + (ghz[r] + bhz))));
        float ng = tanhf((gin[r] + bin) + rg * (ghn[r] + bhn));
        float hv = hf[r * D_ + e];
        snf[r * D_ + e] = (1.f - zg) * ng + zg * hv;
      }
    }
  }
  __syncthreads();

  {  // LN(snf) row `wave` -> sl16 f16
    float xs[8], sm = 0.f, s2 = 0.f;
#pragma unroll
    for (int t = 0; t < 8; ++t) {
      float v = snf[wave * D_ + lane + 32 * t];
      xs[t] = v; sm += v; s2 += v * v;
    }
    sm = wave_reduce_sum(sm); s2 = wave_reduce_sum(s2);
    const float mean = sm * (1.f / D_);
    const float rstd = rsqrtf(s2 * (1.f / D_) - mean * mean + EPS_);
#pragma unroll
    for (int t = 0; t < 8; ++t) {
      const int d = lane + 32 * t;
      sl16[wave * 264 + d] = (_Float16)((xs[t] - mean) * rstd * g_ml[d] + b_ml[d]);
    }
  }
  __syncthreads();

  // ---- h1 = relu(snl @ mw1^T + mb1) ----
  for (int g = wave; g < 16; g += 8) {
    const int eBase = g * 16;
    v8f c0 = {}, c1 = {};
#pragma unroll
    for (int kk = 0; kk < 8; kk += 2) {
      v16h a0 = load_a_frag(sl16, 264, 0, kk * 32, lane);
      v16h b0 = load_b_rm(mw116, eBase, D_, kk * 32, lane);
      c0 = WMMA_F16(a0, b0, c0);
      v16h a1 = load_a_frag(sl16, 264, 0, kk * 32 + 32, lane);
      v16h b1 = load_b_rm(mw116, eBase, D_, kk * 32 + 32, lane);
      c1 = WMMA_F16(a1, b1, c1);
    }
    const int e = eBase + eL;
    const float bb = mb1[e];
    if (lane < 16) {
#pragma unroll
      for (int r = 0; r < 8; ++r)
        h116[r * 264 + e] = (_Float16)fmaxf(c0[r] + c1[r] + bb, 0.f);
    }
  }
  __syncthreads();

  // ---- out = snf + h1 @ mw2^T + mb2 ----
  for (int g = wave; g < 16; g += 8) {
    const int eBase = g * 16;
    v8f c0 = {}, c1 = {};
#pragma unroll
    for (int kk = 0; kk < 8; kk += 2) {
      v16h a0 = load_a_frag(h116, 264, 0, kk * 32, lane);
      v16h b0 = load_b_rm(mw216, eBase, D_, kk * 32, lane);
      c0 = WMMA_F16(a0, b0, c0);
      v16h a1 = load_a_frag(h116, 264, 0, kk * 32 + 32, lane);
      v16h b1 = load_b_rm(mw216, eBase, D_, kk * 32 + 32, lane);
      c1 = WMMA_F16(a1, b1, c1);
    }
    const int e = eBase + eL;
    const float bb = mb2[e];
    if (lane < 16) {
#pragma unroll
      for (int r = 0; r < 8; ++r)
        slots[base + r * D_ + e] = snf[r * D_ + e] + c0[r] + c1[r] + bb;
    }
  }
}

// ---------------------------------------------------------------- launch

extern "C" void kernel_launch(void* const* d_in, const int* in_sizes, int n_in,
                              void* d_out, int out_size, void* d_ws, size_t ws_size,
                              hipStream_t stream) {
  const float* x        = (const float*)d_in[0];
  const float* noise    = (const float*)d_in[1];
  const float* slot_mu  = (const float*)d_in[2];
  const float* slot_sig = (const float*)d_in[3];
  const float* wq   = (const float*)d_in[4];  const float* bq   = (const float*)d_in[5];
  const float* wk   = (const float*)d_in[6];  const float* bk   = (const float*)d_in[7];
  const float* wv   = (const float*)d_in[8];  const float* bv   = (const float*)d_in[9];
  const float* w_ih = (const float*)d_in[10]; const float* w_hh = (const float*)d_in[11];
  const float* b_ih = (const float*)d_in[12]; const float* b_hh = (const float*)d_in[13];
  const float* mw1  = (const float*)d_in[14]; const float* mb1  = (const float*)d_in[15];
  const float* mw2  = (const float*)d_in[16]; const float* mb2  = (const float*)d_in[17];
  const float* g_in = (const float*)d_in[18]; const float* b_in = (const float*)d_in[19];
  const float* g_sl = (const float*)d_in[20]; const float* b_sl = (const float*)d_in[21];
  const float* g_ml = (const float*)d_in[22]; const float* b_ml = (const float*)d_in[23];

  char* ws = (char*)d_ws;
  size_t off = 0;
  _Float16* k16   = (_Float16*)(ws + off); off += (size_t)B_ * N_ * D_ * 2;
  _Float16* vT16  = (_Float16*)(ws + off); off += (size_t)B_ * N_ * D_ * 2;
  float* slots    = (float*)(ws + off);    off += (size_t)B_ * S_ * D_ * 4;
  _Float16* q16   = (_Float16*)(ws + off); off += (size_t)B_ * 16 * D_ * 2;
  float* partial  = (float*)(ws + off);    off += (size_t)B_ * NCHUNK * S_ * D_ * 4;
  float* updates  = (float*)(ws + off);    off += (size_t)B_ * S_ * D_ * 4;
  _Float16* wk16  = (_Float16*)(ws + off); off += (size_t)D_ * D_ * 2;
  _Float16* wv16  = (_Float16*)(ws + off); off += (size_t)D_ * D_ * 2;
  _Float16* wq16  = (_Float16*)(ws + off); off += (size_t)D_ * D_ * 2;
  _Float16* wih16 = (_Float16*)(ws + off); off += (size_t)3 * D_ * D_ * 2;
  _Float16* whh16 = (_Float16*)(ws + off); off += (size_t)3 * D_ * D_ * 2;
  _Float16* mw116 = (_Float16*)(ws + off); off += (size_t)D_ * D_ * 2;
  _Float16* mw216 = (_Float16*)(ws + off); off += (size_t)D_ * D_ * 2;

  const int CT = 256;
  f32_to_f16_kernel<<<dim3((D_ * D_ + CT - 1) / CT), CT, 0, stream>>>(wk, wk16, D_ * D_);
  f32_to_f16_kernel<<<dim3((D_ * D_ + CT - 1) / CT), CT, 0, stream>>>(wv, wv16, D_ * D_);
  f32_to_f16_kernel<<<dim3((D_ * D_ + CT - 1) / CT), CT, 0, stream>>>(wq, wq16, D_ * D_);
  f32_to_f16_kernel<<<dim3((3 * D_ * D_ + CT - 1) / CT), CT, 0, stream>>>(w_ih, wih16, 3 * D_ * D_);
  f32_to_f16_kernel<<<dim3((3 * D_ * D_ + CT - 1) / CT), CT, 0, stream>>>(w_hh, whh16, 3 * D_ * D_);
  f32_to_f16_kernel<<<dim3((D_ * D_ + CT - 1) / CT), CT, 0, stream>>>(mw1, mw116, D_ * D_);
  f32_to_f16_kernel<<<dim3((D_ * D_ + CT - 1) / CT), CT, 0, stream>>>(mw2, mw216, D_ * D_);

  ln_proj_kv_kernel<<<dim3((B_ * N_) / K1_ROWS), 256, 0, stream>>>(
      x, g_in, b_in, wk16, bk, wv16, bv, k16, vT16);
  slots_init_kernel<<<dim3((B_ * S_ * D_ + 255) / 256), 256, 0, stream>>>(
      noise, slot_mu, slot_sig, slots);

  for (int it = 0; it < ITERS_; ++it) {
    q_proj_kernel<<<dim3(B_), 256, 0, stream>>>(slots, g_sl, b_sl, wq16, bq, q16);
    attn_kernel<<<dim3(B_, NCHUNK), 256, 0, stream>>>(q16, k16, vT16, partial);
    reduce_updates_kernel<<<dim3((B_ * S_ * D_ + 255) / 256), 256, 0, stream>>>(
        partial, updates);
    gru_mlp_kernel<<<dim3(B_), 256, 0, stream>>>(slots, updates, wih16, whh16, b_ih,
                                                 b_hh, mw116, mb1, mw216, mb2, g_ml, b_ml);
  }
  hipMemcpyAsync(d_out, slots, (size_t)B_ * S_ * D_ * sizeof(float),
                 hipMemcpyDeviceToDevice, stream);
}
